// NLinear_40166534152349
// MI455X (gfx1250) — compile-verified, matrix operand order checked by reading, not run on hardware
//
#include <hip/hip_runtime.h>

// ---------------------------------------------------------------------------
// NLinear (per-position batched GEMM) for gfx1250, split-bf16 WMMA,
// double-buffered LDS pipeline.
// out[b,n,o] = sum_i x[b,n,i] * W[n,i,o] + bias[n,o]
//   x: (256,128,512) f32   W: (128,512,512) f32   bias: (128,512) f32
// ---------------------------------------------------------------------------

typedef __attribute__((ext_vector_type(16))) __bf16 v16bf;
typedef __attribute__((ext_vector_type(8)))  float  v8f;

#define NPOS 128   // positions (independent GEMMs)
#define MB   256   // batch rows (GEMM M)
#define KD   512   // D_IN  (GEMM K)
#define ND   512   // D_OUT (GEMM N)

#define BM 128     // block tile M
#define BN 128     // block tile N
#define BK 32      // K-step (one bf16 WMMA K slice)
#define NT (KD / BK)
#define LSTR 40    // padded LDS row stride in bf16 elems (80 B)

// LDS layout (in unsigned short units), double buffered:
#define TSZ   (128 * LSTR)        // one tile array: 5120 shorts (10240 B)
#define OFF_AHI 0
#define OFF_ALO (1 * TSZ)
#define OFF_BHI (2 * TSZ)
#define OFF_BLO (3 * TSZ)
#define BUFS    (4 * TSZ)         // 20480 shorts per buffer
#define SMEM_BYTES (2 * BUFS * 2) // 81920 bytes

union FragU { uint4 u[2]; v16bf v; };

// WMMA 16-bit A/B fragment from an LDS tile with LSTR-short rows:
//   lane<16 : row = r, K byte runs {0..15} and {32..47}
//   lane>=16: row = r, runs shifted by +16 B
__device__ __forceinline__ v16bf load_frag(const unsigned short* tile, int row0,
                                           int r, int half) {
    const unsigned short* p = tile + (row0 + r) * LSTR + half * 8;
    FragU f;
    f.u[0] = *(const uint4*)(p);
    f.u[1] = *(const uint4*)(p + 16);
    return f.v;
}

__device__ __forceinline__ void load_a_regs(const float* xg, int aRow, int aKc,
                                            int k0, float* a) {
    const float* s = xg + (size_t)aRow * (NPOS * KD) + k0 + aKc;
#pragma unroll
    for (int j = 0; j < 4; ++j) {
        float4 v = *(const float4*)(s + 4 * j);
        a[4 * j + 0] = v.x; a[4 * j + 1] = v.y;
        a[4 * j + 2] = v.z; a[4 * j + 3] = v.w;
    }
}

__device__ __forceinline__ void load_b_regs(const float* wg, int bCol, int bKs,
                                            int k0, float* b) {
    const float* s = wg + (size_t)(k0 + bKs) * ND + bCol;
#pragma unroll
    for (int j = 0; j < 16; ++j) b[j] = s[(size_t)j * ND];
}

// Truncation split of 16 fp32 into packed bf16 hi/lo, stored as 2x uint4 each.
// hi = top 16 bits of f; lo = bf16_trunc(f - hi).  v_perm_b32 packs pairs.
__device__ __forceinline__ void store_split16(unsigned short* dh,
                                              unsigned short* dl,
                                              const float* f) {
    unsigned hu[8], lu[8];
#pragma unroll
    for (int p = 0; p < 8; ++p) {
        unsigned u0 = __float_as_uint(f[2 * p + 0]);
        unsigned u1 = __float_as_uint(f[2 * p + 1]);
        hu[p] = __builtin_amdgcn_perm(u1, u0, 0x07060302u);
        float l0 = f[2 * p + 0] - __uint_as_float(u0 & 0xFFFF0000u);
        float l1 = f[2 * p + 1] - __uint_as_float(u1 & 0xFFFF0000u);
        lu[p] = __builtin_amdgcn_perm(__float_as_uint(l1), __float_as_uint(l0),
                                      0x07060302u);
    }
    *(uint4*)(dh)     = make_uint4(hu[0], hu[1], hu[2], hu[3]);
    *(uint4*)(dh + 8) = make_uint4(hu[4], hu[5], hu[6], hu[7]);
    *(uint4*)(dl)     = make_uint4(lu[0], lu[1], lu[2], lu[3]);
    *(uint4*)(dl + 8) = make_uint4(lu[4], lu[5], lu[6], lu[7]);
}

__global__ __launch_bounds__(256)
void nlinear_wmma_kernel(const float* __restrict__ x,
                         const float* __restrict__ w,
                         const float* __restrict__ bias,
                         float* __restrict__ out)
{
    extern __shared__ unsigned short smem[];

    const int bid  = blockIdx.x;
    const int npos = bid >> 3;          // 128 positions
    const int bM   = (bid >> 2) & 1;    // 2 M tiles
    const int bN   = bid & 3;           // 4 N tiles

    const int tid  = threadIdx.x;
    const int lane = tid & 31;
    const int wave = tid >> 5;          // 0..7
    const int waveM = wave & 3;         // 4 waves along M -> 32 rows each
    const int waveN = wave >> 2;        // 2 waves along N -> 64 cols each
    const int r    = lane & 15;
    const int half = lane >> 4;

    const float* xg = x + (size_t)npos * KD + (size_t)(bM * BM) * (NPOS * KD);
    const float* wg = w + (size_t)npos * KD * ND + (size_t)(bN * BN);

    // fill-phase mapping (all 256 threads):
    const int aRow = tid >> 1;          // A: 2 threads/row, 16 floats each
    const int aKc  = (tid & 1) * 16;
    const int bCol = tid & 127;         // B: thread owns column n, 16 k values
    const int bKs  = (tid >> 7) * 16;

    v8f acc[2][4];
#pragma unroll
    for (int mi = 0; mi < 2; ++mi)
#pragma unroll
        for (int ni = 0; ni < 4; ++ni)
            acc[mi][ni] = (v8f)(0.0f);

    float aReg[16], bReg[16];

    // ---- prologue: stage K-step 0 into buffer 0 ----
    load_a_regs(xg, aRow, aKc, 0, aReg);
    load_b_regs(wg, bCol, bKs, 0, bReg);
    {
        unsigned short* nb = smem;  // buffer 0
        store_split16(nb + OFF_AHI + aRow * LSTR + aKc,
                      nb + OFF_ALO + aRow * LSTR + aKc, aReg);
        store_split16(nb + OFF_BHI + bCol * LSTR + bKs,
                      nb + OFF_BLO + bCol * LSTR + bKs, bReg);
    }
    __syncthreads();

    for (int t = 0; t < NT; ++t) {
        // ---- issue next tile's global loads early (hidden under WMMA) ----
        if (t + 1 < NT) {
            const int k1 = (t + 1) * BK;
            load_a_regs(xg, aRow, aKc, k1, aReg);
            load_b_regs(wg, bCol, bKs, k1, bReg);
            if (t + 2 < NT) {
                __builtin_prefetch(xg + (size_t)aRow * (NPOS * KD) + k1 + BK + aKc, 0, 1);
                __builtin_prefetch(wg + (size_t)(k1 + BK + bKs) * ND + bCol, 0, 1);
            }
        }

        // ---- compute on buffer t&1: 24 WMMAs ----
        const unsigned short* buf = smem + (t & 1) * BUFS;
        v16bf aHi[2], aLo[2], bHi[4], bLo[4];
#pragma unroll
        for (int mi = 0; mi < 2; ++mi) {
            aHi[mi] = load_frag(buf + OFF_AHI, waveM * 32 + mi * 16, r, half);
            aLo[mi] = load_frag(buf + OFF_ALO, waveM * 32 + mi * 16, r, half);
        }
#pragma unroll
        for (int ni = 0; ni < 4; ++ni) {
            bHi[ni] = load_frag(buf + OFF_BHI, waveN * 64 + ni * 16, r, half);
            bLo[ni] = load_frag(buf + OFF_BLO, waveN * 64 + ni * 16, r, half);
        }
#pragma unroll
        for (int mi = 0; mi < 2; ++mi) {
#pragma unroll
            for (int ni = 0; ni < 4; ++ni) {
                v8f c = acc[mi][ni];
                c = __builtin_amdgcn_wmma_f32_16x16x32_bf16(
                        false, aHi[mi], false, bHi[ni], (short)0, c, false, false);
                c = __builtin_amdgcn_wmma_f32_16x16x32_bf16(
                        false, aHi[mi], false, bLo[ni], (short)0, c, false, false);
                c = __builtin_amdgcn_wmma_f32_16x16x32_bf16(
                        false, aLo[mi], false, bHi[ni], (short)0, c, false, false);
                acc[mi][ni] = c;
            }
        }

        // ---- convert + store next tile into the other buffer ----
        if (t + 1 < NT) {
            unsigned short* nb = smem + ((t + 1) & 1) * BUFS;
            store_split16(nb + OFF_AHI + aRow * LSTR + aKc,
                          nb + OFF_ALO + aRow * LSTR + aKc, aReg);
            store_split16(nb + OFF_BHI + bCol * LSTR + bKs,
                          nb + OFF_BLO + bCol * LSTR + bKs, bReg);
        }
        __syncthreads();
    }

    // ---- epilogue: add bias, store fp32 ----
    // C/D layout: VGPR j -> M = j + 8*half, N = r (within a 16x16 tile)
    const int oBase = bN * BN + waveN * 64;
    const float* bp = bias + (size_t)npos * ND;
    float* og = out + (size_t)npos * ND + (size_t)(bM * BM) * (NPOS * ND);
#pragma unroll
    for (int ni = 0; ni < 4; ++ni) {
        const int o  = oBase + ni * 16 + r;
        const float bv = bp[o];
#pragma unroll
        for (int mi = 0; mi < 2; ++mi) {
            const int m0 = waveM * 32 + mi * 16 + half * 8;
#pragma unroll
            for (int j = 0; j < 8; ++j) {
                og[(size_t)(m0 + j) * (NPOS * ND) + o] = acc[mi][ni][j] + bv;
            }
        }
    }
}

extern "C" void kernel_launch(void* const* d_in, const int* in_sizes, int n_in,
                              void* d_out, int out_size, void* d_ws, size_t ws_size,
                              hipStream_t stream) {
    (void)in_sizes; (void)n_in; (void)out_size; (void)d_ws; (void)ws_size;
    const float* x  = (const float*)d_in[0];
    const float* w  = (const float*)d_in[1];
    const float* bi = (const float*)d_in[2];
    float* out = (float*)d_out;

    static bool attr_done = false;  // idempotent, deterministic (same every call)
    if (!attr_done) {
        hipFuncSetAttribute((const void*)nlinear_wmma_kernel,
                            hipFuncAttributeMaxDynamicSharedMemorySize,
                            SMEM_BYTES);
        attr_done = true;
    }

    // grid: 128 positions x (256/128) M-tiles x (512/128) N-tiles = 1024 blocks
    dim3 grid(NPOS * (MB / BM) * (ND / BN));
    nlinear_wmma_kernel<<<grid, 256, SMEM_BYTES, stream>>>(x, w, bi, out);
}
